// GRU_13460427506121
// MI455X (gfx1250) — matrix operand order repaired and checked by reading, not compiled, not run on hardware
//
#include <hip/hip_runtime.h>
#include <cstdint>
#include <cstddef>

#define NB   16      // batch
#define SEQ  1024    // sequence
#define DM   512     // model dim
#define NH   8       // heads
#define LS   32      // decode steps
#define DHD  64      // head dim
#define G3   (3*DM)  // 1536 (GRU gates)
#define XD   (DM+3)  // 515

typedef unsigned short u16;
typedef __attribute__((ext_vector_type(16))) __bf16 v16bf;
typedef __attribute__((ext_vector_type(8)))  __bf16 v8bf;
typedef __attribute__((ext_vector_type(8)))  float  v8f;

static __device__ __forceinline__ float bf2f(u16 u) {
    unsigned int x = ((unsigned int)u) << 16;
    return __builtin_bit_cast(float, x);
}
static __device__ __forceinline__ u16 f2bf(float f) {
    unsigned int u = __builtin_bit_cast(unsigned int, f);
    unsigned int r = (u + 0x7FFFu + ((u >> 16) & 1u)) >> 16;
    return (u16)r;
}
// Generic shared pointer = {shared_aperture_hi32, lds_byte_offset}; truncation
// recovers the LDS byte address (same lowering as addrspacecast AS0->AS3).
static __device__ __forceinline__ unsigned lds_addr32(const void* p) {
    return (unsigned)(uintptr_t)p;
}

// ---------------------------------------------------------------- converts
__global__ void k_f32_to_bf16(const float* __restrict__ src, u16* __restrict__ dst, int n) {
    int i = blockIdx.x * blockDim.x + threadIdx.x;
    if (i < n) dst[i] = f2bf(src[i]);
}

__global__ void k_init_h(const float* __restrict__ e_last, float* __restrict__ h,
                         u16* __restrict__ hbf) {
    int i = blockIdx.x * blockDim.x + threadIdx.x;
    if (i < NB * DM) { float v = e_last[i]; h[i] = v; hbf[i] = f2bf(v); }
}

// ------------------------------------------------- big GEMM: [16384,512]x[512,512]^T
// One workgroup (8 waves) per 16-row M strip. The contiguous 32KB f32 A strip is
// async-copied to LDS once (GLOBAL_LOAD_ASYNC_TO_LDS_B128, ASYNCcnt), converted
// to bf16 once, then each wave computes a 16x64 output group: A fragments come
// from LDS, B (bf16 weight, native [N,K] layout) streams from global/L2.
__global__ void __launch_bounds__(256)
k_gemm_big(const float* __restrict__ A, const u16* __restrict__ W,
           const float* __restrict__ bias, u16* __restrict__ out) {
    __shared__ float Af[16 * DM];   // 32 KB staged f32 strip
    __shared__ u16   Ab[16 * DM];   // 16 KB bf16 strip

    const int t  = threadIdx.x;     // 0..255
    const int mt = blockIdx.x;      // 0..1023 (M tile)

    // ---- async stage: 32 KB contiguous, 16 B per lane per op, 8 rounds
    {
        const char*    gbase = (const char*)(A + (size_t)mt * 16 * DM);
        const unsigned lbase = lds_addr32(Af);
#pragma unroll
        for (int r = 0; r < 8; ++r) {
            const unsigned byteOff = (unsigned)(r * 256 + t) * 16u;
            const unsigned ldst    = lbase + byteOff;
            const unsigned long long gaddr = (unsigned long long)(gbase + byteOff);
            asm volatile("global_load_async_to_lds_b128 %0, %1, off"
                         :: "v"(ldst), "v"(gaddr) : "memory");
        }
        asm volatile("s_wait_asynccnt 0x0" ::: "memory");
    }
    __syncthreads();

    // ---- convert the strip to bf16 once for all 8 waves
#pragma unroll
    for (int r = 0; r < 32; ++r) {
        const int i = r * 256 + t;
        Ab[i] = f2bf(Af[i]);
    }
    __syncthreads();

    // ---- per-wave 16x64 WMMA strip
    const int lane = t & 31;
    const int l    = lane & 15;
    const int hi   = lane >> 4;
    const int N0   = (t >> 5) * 64;         // wave id * 64 cols

    const u16* xrow = Ab + (size_t)l * DM;
    v8f acc0 = {}, acc1 = {}, acc2 = {}, acc3 = {};

    for (int kc = 0; kc < DM; kc += 32) {
        const int cb = kc + hi * 16;
        v8bf a0 = *(const v8bf*)(xrow + cb);
        v8bf a1 = *(const v8bf*)(xrow + cb + 8);
        v16bf a;
#pragma unroll
        for (int i = 0; i < 8; ++i) { a[i] = a0[i]; a[8 + i] = a1[i]; }

        const u16* wb = W + (size_t)(N0 + l) * DM + kc + hi * 16;
        v16bf b0 = *(const v16bf*)(wb + 0 * 16 * DM);
        v16bf b1 = *(const v16bf*)(wb + 1 * 16 * DM);
        v16bf b2 = *(const v16bf*)(wb + 2 * 16 * DM);
        v16bf b3 = *(const v16bf*)(wb + 3 * 16 * DM);

        acc0 = __builtin_amdgcn_wmma_f32_16x16x32_bf16(false, a, false, b0, (short)0, acc0, false, false);
        acc1 = __builtin_amdgcn_wmma_f32_16x16x32_bf16(false, a, false, b1, (short)0, acc1, false, false);
        acc2 = __builtin_amdgcn_wmma_f32_16x16x32_bf16(false, a, false, b2, (short)0, acc2, false, false);
        acc3 = __builtin_amdgcn_wmma_f32_16x16x32_bf16(false, a, false, b3, (short)0, acc3, false, false);
    }

    const float bv0 = bias[N0 + 0  + l];
    const float bv1 = bias[N0 + 16 + l];
    const float bv2 = bias[N0 + 32 + l];
    const float bv3 = bias[N0 + 48 + l];
#pragma unroll
    for (int r = 0; r < 8; ++r) {
        const size_t m = (size_t)(mt * 16 + hi * 8 + r);
        out[m * DM + N0 + 0  + l] = f2bf(acc0[r] + bv0);
        out[m * DM + N0 + 16 + l] = f2bf(acc1[r] + bv1);
        out[m * DM + N0 + 32 + l] = f2bf(acc2[r] + bv2);
        out[m * DM + N0 + 48 + l] = f2bf(acc3[r] + bv3);
    }
}

// ------------------------------------------------- small GEMM: [16,512]x[512,512]^T
// One wave per 16x16 output tile; M=16 == batch. X is bf16, out is f32 (+bias).
__global__ void __launch_bounds__(32)
k_gemm16(const u16* __restrict__ X, const u16* __restrict__ W,
         const float* __restrict__ bias, float* __restrict__ out) {
    const int lane = threadIdx.x;
    const int l  = lane & 15;
    const int hi = lane >> 4;
    const int N0 = blockIdx.x * 16;

    const u16* xrow = X + (size_t)l * DM;
    v8f acc = {};
    for (int kc = 0; kc < DM; kc += 32) {
        const int cb = kc + hi * 16;
        v8bf a0 = *(const v8bf*)(xrow + cb);
        v8bf a1 = *(const v8bf*)(xrow + cb + 8);
        v16bf a;
#pragma unroll
        for (int i = 0; i < 8; ++i) { a[i] = a0[i]; a[8 + i] = a1[i]; }
        v16bf b = *(const v16bf*)(W + (size_t)(N0 + l) * DM + kc + hi * 16);
        acc = __builtin_amdgcn_wmma_f32_16x16x32_bf16(false, a, false, b, (short)0, acc, false, false);
    }
    const float bv = bias[N0 + l];
#pragma unroll
    for (int r = 0; r < 8; ++r)
        out[(size_t)(hi * 8 + r) * DM + N0 + l] = acc[r] + bv;
}

// ------------------------------------------------- attention for one step
// One block per (b,h). scores = K_h[1024,64] @ q_h / 32 ; softmax ; attn = dist @ V_h.
__global__ void __launch_bounds__(256)
k_attn(const u16* __restrict__ Kbf, const u16* __restrict__ Vbf,
       const float* __restrict__ qf, u16* __restrict__ attn_bf,
       float* __restrict__ cross, int step) {
    const int b = blockIdx.x >> 3;
    const int h = blockIdx.x & 7;
    const int t = threadIdx.x;

    __shared__ float qs[DHD];
    __shared__ float red[256];
    __shared__ float dist[SEQ];

    const u16* Kh = Kbf + (size_t)b * SEQ * DM + (size_t)h * SEQ * DHD;
    const u16* Vh = Vbf + (size_t)b * SEQ * DM + (size_t)h * SEQ * DHD;

    if (t < DHD) qs[t] = qf[b * DM + h * DHD + t];
    __syncthreads();

    float sc[4];
    float lm = -3.4e38f;
#pragma unroll
    for (int j = 0; j < 4; ++j) {
        const int k = j * 256 + t;
        const u16* kr = Kh + (size_t)k * DHD;
        float d = 0.f;
#pragma unroll
        for (int dh = 0; dh < DHD; ++dh) d += qs[dh] * bf2f(kr[dh]);
        sc[j] = d * (1.0f / 32.0f);           // /(DH/2)
        lm = fmaxf(lm, sc[j]);
    }
    red[t] = lm;
    __syncthreads();
    for (int s = 128; s > 0; s >>= 1) {
        if (t < s) red[t] = fmaxf(red[t], red[t + s]);
        __syncthreads();
    }
    const float m = red[0];
    __syncthreads();

    float ls = 0.f;
#pragma unroll
    for (int j = 0; j < 4; ++j) {
        float e = expf(sc[j] - m);
        dist[j * 256 + t] = e;
        ls += e;
    }
    red[t] = ls;
    __syncthreads();
    for (int s = 128; s > 0; s >>= 1) {
        if (t < s) red[t] += red[t + s];
        __syncthreads();
    }
    const float inv = 1.0f / red[0];
    __syncthreads();

    float* crow = cross + (((size_t)(b * NH + h)) * LS + step) * SEQ;
#pragma unroll
    for (int j = 0; j < 4; ++j) {
        const int k = j * 256 + t;
        float p = dist[k] * inv;
        dist[k] = p;
        crow[k] = p;
    }
    __syncthreads();

    // attn[dh] = sum_k dist[k] * V_h[k][dh]; 4 partial sums per dh
    const int dh   = t & 63;
    const int part = t >> 6;
    float acc = 0.f;
    const int k0 = part * 256;
#pragma unroll 8
    for (int k = k0; k < k0 + 256; ++k) acc += dist[k] * bf2f(Vh[(size_t)k * DHD + dh]);
    red[t] = acc;
    __syncthreads();
    if (part == 0) {
        float a = red[dh] + red[dh + 64] + red[dh + 128] + red[dh + 192];
        attn_bf[b * DM + h * DHD + dh] = f2bf(a);
    }
}

// ------------------------------------------------- GRU gate pre-activations
__global__ void __launch_bounds__(256)
k_gates(const float* __restrict__ attnO, const float* __restrict__ h,
        const float* __restrict__ target,
        const float* __restrict__ W_ih, const float* __restrict__ W_hh,
        const float* __restrict__ b_ih, const float* __restrict__ b_hh,
        float* __restrict__ gi, float* __restrict__ gh, int step) {
    const int idx = blockIdx.x * blockDim.x + threadIdx.x;
    if (idx >= NB * G3) return;
    const int b = idx / G3;
    const int j = idx % G3;

    const float* x   = attnO + (size_t)b * DM;
    const float* wih = W_ih + (size_t)j * XD;
    float si = b_ih[j];
    for (int k = 0; k < DM; ++k) si += x[k] * wih[k];
    float d0 = 0.f, d1 = 0.f, d2 = 0.f;
    if (step > 0) {
        const float* tg = target + (size_t)b * (LS * 3) + (size_t)(step - 1) * 3;
        d0 = tg[0]; d1 = tg[1]; d2 = tg[2];
    }
    si += d0 * wih[DM] + d1 * wih[DM + 1] + d2 * wih[DM + 2];

    const float* hb  = h + (size_t)b * DM;
    const float* whh = W_hh + (size_t)j * DM;
    float sh = b_hh[j];
    for (int k = 0; k < DM; ++k) sh += hb[k] * whh[k];

    gi[idx] = si;
    gh[idx] = sh;
}

// ------------------------------------------------- GRU state update
__global__ void __launch_bounds__(256)
k_update(const float* __restrict__ gi, const float* __restrict__ gh,
         float* __restrict__ h, u16* __restrict__ hbf,
         float* __restrict__ hfin, int writeFin) {
    const int idx = blockIdx.x * blockDim.x + threadIdx.x;
    if (idx >= NB * DM) return;
    const int b = idx >> 9;
    const int d = idx & 511;
    const size_t base = (size_t)b * G3;

    const float i_r = gi[base + d],          h_r = gh[base + d];
    const float i_z = gi[base + DM + d],     h_z = gh[base + DM + d];
    const float i_n = gi[base + 2 * DM + d], h_n = gh[base + 2 * DM + d];

    const float r = 1.0f / (1.0f + expf(-(i_r + h_r)));
    const float z = 1.0f / (1.0f + expf(-(i_z + h_z)));
    const float n = tanhf(i_n + r * h_n);
    const float hold = h[idx];
    const float hn = (1.0f - z) * n + z * hold;

    h[idx]   = hn;
    hbf[idx] = f2bf(hn);
    if (writeFin) hfin[idx] = hn;
}

// ------------------------------------------------- output projection [B,3]
__global__ void k_outproj(const float* __restrict__ h, const float* __restrict__ W_out,
                          const float* __restrict__ b_out, float* __restrict__ dout, int step) {
    const int t = threadIdx.x;
    if (t >= NB * 3) return;
    const int b = t / 3, c = t % 3;
    const float* hr = h + (size_t)b * DM;
    const float* wr = W_out + (size_t)c * DM;
    float s = b_out[c];
    for (int k = 0; k < DM; ++k) s += hr[k] * wr[k];
    dout[(size_t)b * (LS * 3) + (size_t)step * 3 + c] = s;
}

// ----------------------------------------------------------------- launcher
extern "C" void kernel_launch(void* const* d_in, const int* in_sizes, int n_in,
                              void* d_out, int out_size, void* d_ws, size_t ws_size,
                              hipStream_t stream) {
    (void)in_sizes; (void)n_in; (void)out_size; (void)ws_size;

    const float* e_all  = (const float*)d_in[0];
    const float* e_last = (const float*)d_in[1];
    const float* target = (const float*)d_in[2];
    const float* Wq = (const float*)d_in[3];  const float* bq = (const float*)d_in[4];
    const float* Wk = (const float*)d_in[5];  const float* bk = (const float*)d_in[6];
    const float* Wv = (const float*)d_in[7];  const float* bv = (const float*)d_in[8];
    const float* Wo = (const float*)d_in[9];  const float* bo = (const float*)d_in[10];
    const float* W_ih = (const float*)d_in[11]; const float* W_hh = (const float*)d_in[12];
    const float* b_ih = (const float*)d_in[13]; const float* b_hh = (const float*)d_in[14];
    const float* W_out = (const float*)d_in[15]; const float* b_out = (const float*)d_in[16];

    float* out        = (float*)d_out;
    float* d_outputs  = out;                        // [B,L,3]   1536
    float* h_fin      = out + NB * LS * 3;          // [1,B,D]   8192
    float* cross      = h_fin + NB * DM;            // [B,H,L,S]

    // workspace carve (256B aligned)
    char* wp = (char*)d_ws;
    auto carve = [&](size_t bytes) -> char* {
        char* p = wp; wp += (bytes + 255) & ~(size_t)255; return p;
    };
    u16*   Kbf     = (u16*)carve((size_t)NB * SEQ * DM * sizeof(u16));
    u16*   Vbf     = (u16*)carve((size_t)NB * SEQ * DM * sizeof(u16));
    u16*   Wqb     = (u16*)carve((size_t)DM * DM * sizeof(u16));
    u16*   Wkb     = (u16*)carve((size_t)DM * DM * sizeof(u16));
    u16*   Wvb     = (u16*)carve((size_t)DM * DM * sizeof(u16));
    u16*   Wob     = (u16*)carve((size_t)DM * DM * sizeof(u16));
    float* qf      = (float*)carve((size_t)NB * DM * sizeof(float));
    u16*   attn_bf = (u16*)carve((size_t)NB * DM * sizeof(u16));
    float* attnO   = (float*)carve((size_t)NB * DM * sizeof(float));
    float* gi      = (float*)carve((size_t)NB * G3 * sizeof(float));
    float* gh      = (float*)carve((size_t)NB * G3 * sizeof(float));
    float* hbuf    = (float*)carve((size_t)NB * DM * sizeof(float));
    u16*   hbf     = (u16*)carve((size_t)NB * DM * sizeof(u16));

    const int nW = DM * DM;
    k_f32_to_bf16<<<(nW + 255) / 256, 256, 0, stream>>>(Wq, Wqb, nW);
    k_f32_to_bf16<<<(nW + 255) / 256, 256, 0, stream>>>(Wk, Wkb, nW);
    k_f32_to_bf16<<<(nW + 255) / 256, 256, 0, stream>>>(Wv, Wvb, nW);
    k_f32_to_bf16<<<(nW + 255) / 256, 256, 0, stream>>>(Wo, Wob, nW);
    k_init_h<<<(NB * DM + 255) / 256, 256, 0, stream>>>(e_last, hbuf, hbf);

    // hoisted K/V projections: 1024 workgroups, 8 waves each (async-LDS staged A)
    k_gemm_big<<<1024, 256, 0, stream>>>(e_all, Wkb, bk, Kbf);
    k_gemm_big<<<1024, 256, 0, stream>>>(e_all, Wvb, bv, Vbf);

    for (int step = 0; step < LS; ++step) {
        k_gemm16<<<DM / 16, 32, 0, stream>>>(hbf, Wqb, bq, qf);                     // q = h.Wq^T+bq
        k_attn<<<NB * NH, 256, 0, stream>>>(Kbf, Vbf, qf, attn_bf, cross, step);    // softmax attn
        k_gemm16<<<DM / 16, 32, 0, stream>>>(attn_bf, Wob, bo, attnO);              // Wo proj
        k_gates<<<(NB * G3) / 256, 256, 0, stream>>>(attnO, hbuf, target, W_ih, W_hh,
                                                     b_ih, b_hh, gi, gh, step);
        k_update<<<(NB * DM) / 256, 256, 0, stream>>>(gi, gh, hbuf, hbf, h_fin,
                                                      step == LS - 1);
        k_outproj<<<1, 64, 0, stream>>>(hbuf, W_out, b_out, d_outputs, step);
    }
}
